// RelativeAttention_60825326846014
// MI455X (gfx1250) — compile-verified, matrix operand order checked by reading, not run on hardware
//
#include <hip/hip_runtime.h>
#include <cstdint>
#include <cstddef>

// ---------------------------------------------------------------------------
// RelativeAttention (ALiBi bidirectional) for MI455X / gfx1250, wave32 WMMA.
// All matmuls run on v_wmma_f32_16x16x32_f16 (f16 in, f32 accumulate).
// Weights are pre-transposed (and V written transposed) so every LDS staging
// access is a 128-bit load/store -- no 16-bit scatter in the hot loops.
// ---------------------------------------------------------------------------

typedef _Float16 half_t;
typedef __attribute__((ext_vector_type(16))) _Float16 v16h;
typedef __attribute__((ext_vector_type(8)))  _Float16 v8h;
typedef __attribute__((ext_vector_type(4)))  _Float16 v4h;
typedef __attribute__((ext_vector_type(8)))  float    v8f;

union V16 { v16h v; v8h h[2]; };

constexpr int B_   = 2;
constexpr int S_   = 2048;
constexpr int D_   = 1024;
constexpr int H_   = 16;
constexpr int HD_  = 64;
constexpr int ROWS = B_ * S_;   // 4096

// ---------------------------------------------------------------------------
// f32 -> f16 conversion (vectorized, grid-stride)
// ---------------------------------------------------------------------------
__global__ void cvt_f32_to_f16(const float* __restrict__ src,
                               half_t* __restrict__ dst, int n4) {
  int i = blockIdx.x * blockDim.x + threadIdx.x;
  int stride = gridDim.x * blockDim.x;
  for (; i < n4; i += stride) {
    float4 f = ((const float4*)src)[i];
    v4h h;
    h[0] = (half_t)f.x; h[1] = (half_t)f.y;
    h[2] = (half_t)f.z; h[3] = (half_t)f.w;
    ((v4h*)dst)[i] = h;
  }
}

// ---------------------------------------------------------------------------
// f32 [K][N] -> f16 [N][K] tiled transpose-convert (for the weight matrices).
// 64x64 tiles through LDS, coalesced on both sides. Grid (16,16), 256 thr.
// ---------------------------------------------------------------------------
__global__ void cvt_transpose_f32_to_f16(const float* __restrict__ src,
                                         half_t* __restrict__ dst) {
  __shared__ float tile[64][65];
  const int nb = blockIdx.x * 64;   // n base (cols of src, rows of dst)
  const int kb = blockIdx.y * 64;   // k base (rows of src, cols of dst)
#pragma unroll
  for (int i = 0; i < 16; ++i) {
    int idx = threadIdx.x + i * 256;
    int r = idx >> 6, c = idx & 63;
    tile[r][c] = src[(size_t)(kb + r) * D_ + nb + c];
  }
  __syncthreads();
#pragma unroll
  for (int i = 0; i < 16; ++i) {
    int idx = threadIdx.x + i * 256;
    int r = idx >> 6, c = idx & 63;
    dst[(size_t)(nb + r) * D_ + kb + c] = (half_t)tile[c][r];
  }
}

// ---------------------------------------------------------------------------
// Shared WMMA GEMM core: C(64x128 per block) = A(ROWS x 1024) * B(1024 x 1024)
// 256 threads = 8 waves, each wave computes a 32x32 patch (2x2 WMMA tiles).
// A staged row-major; B comes from a PRE-TRANSPOSED weight Wt[n][k], so both
// stagings and all fragment fills are contiguous 128-bit LDS operations.
// Fragment layout (ISA 7.12.2, 16-bit, wave32):
//   A: lane(l&15)=row, halves at k = {hi*8..+7, 16+hi*8..+7}
//   B: lane(l&15)=col, same k pattern.  C: n=l&15, m=r+8*(l>>4).
// ---------------------------------------------------------------------------
__device__ __forceinline__ void gemm_tile_core(
    const half_t* __restrict__ A, const half_t* __restrict__ Bt_g,
    half_t (*As)[48], half_t (*Bt)[48],
    int rowBase, int colBase, v8f acc[2][2]) {
  const int tid  = threadIdx.x;
  const int lane = tid & 31;
  const int wid  = tid >> 5;
  const int wm   = wid >> 2;       // 0..1  (m sub-block of 32 rows)
  const int wn   = wid & 3;        // 0..3  (n sub-block of 32 cols)
  const int hi   = lane >> 4;      // half-wave select
  const int l15  = lane & 15;

  for (int kt = 0; kt < D_ / 32; ++kt) {
    const int kb = kt * 32;
    // ---- stage A tile 64x32 (row-major), one b128 per thread -------------
    {
      int row = tid >> 2, cb = (tid & 3) * 8;
      v8h t = *(const v8h*)&A[(size_t)(rowBase + row) * D_ + kb + cb];
      *(v8h*)&As[row][cb] = t;
    }
    // ---- stage B tile from Wt[n][k]: 128 n x 32 k, two b128 per thread ---
    {
      int n = tid >> 1, k16 = (tid & 1) * 16;
      const half_t* src = &Bt_g[(size_t)(colBase + n) * D_ + kb + k16];
      v8h lo = *(const v8h*)src;
      v8h hi8 = *(const v8h*)(src + 8);
      *(v8h*)&Bt[n][k16]     = lo;
      *(v8h*)&Bt[n][k16 + 8] = hi8;
    }
    // prefetch next A tile row (speculative)
    if (kt + 1 < D_ / 32) {
      __builtin_prefetch(&A[(size_t)(rowBase + (tid >> 2)) * D_ + kb + 32], 0, 1);
    }
    __syncthreads();

    V16 af[2], bf2[2];
#pragma unroll
    for (int sm = 0; sm < 2; ++sm) {
      int r = wm * 32 + sm * 16 + l15;
      af[sm].h[0] = *(const v8h*)&As[r][hi * 8];
      af[sm].h[1] = *(const v8h*)&As[r][16 + hi * 8];
    }
#pragma unroll
    for (int sn = 0; sn < 2; ++sn) {
      int c = wn * 32 + sn * 16 + l15;
      bf2[sn].h[0] = *(const v8h*)&Bt[c][hi * 8];
      bf2[sn].h[1] = *(const v8h*)&Bt[c][16 + hi * 8];
    }
#pragma unroll
    for (int sm = 0; sm < 2; ++sm)
#pragma unroll
      for (int sn = 0; sn < 2; ++sn)
        acc[sm][sn] = __builtin_amdgcn_wmma_f32_16x16x32_f16(
            false, af[sm].v, false, bf2[sn].v, (short)0, acc[sm][sn],
            false, false);
    __syncthreads();
  }
}

// ---------------------------------------------------------------------------
// Fused QKV projection: z=0 -> Q (pre-scaled by 1/sqrt(hd)), z=1 -> K,
// z=2 -> V written TRANSPOSED [B, H, hd, S] so the attention kernel can
// stage V tiles with contiguous b128 accesses. Q/K go to [B, H, S, hd].
// ---------------------------------------------------------------------------
__global__ void qkv_gemm_kernel(
    const half_t* __restrict__ Xh,
    const half_t* __restrict__ Wqt, const half_t* __restrict__ Wkt,
    const half_t* __restrict__ Wvt,
    const float* __restrict__ bq, const float* __restrict__ bk,
    const float* __restrict__ bv,
    half_t* __restrict__ Qh, half_t* __restrict__ Kh,
    half_t* __restrict__ VhT) {
  __shared__ __align__(16) half_t As[64][48];
  __shared__ __align__(16) half_t Bt[128][48];

  const half_t* W; const float* bias; half_t* dst; float scale;
  bool vmode = false;
  if (blockIdx.z == 0)      { W = Wqt; bias = bq; dst = Qh;  scale = 0.125f; }
  else if (blockIdx.z == 1) { W = Wkt; bias = bk; dst = Kh;  scale = 1.0f; }
  else                      { W = Wvt; bias = bv; dst = VhT; scale = 1.0f; vmode = true; }

  const int rowBase = blockIdx.y * 64;
  const int colBase = blockIdx.x * 128;
  v8f acc[2][2] = {};
  gemm_tile_core(Xh, W, As, Bt, rowBase, colBase, acc);

  const int lane = threadIdx.x & 31;
  const int wid  = threadIdx.x >> 5;
  const int wm = wid >> 2, wn = wid & 3, hi = lane >> 4, l15 = lane & 15;
#pragma unroll
  for (int sm = 0; sm < 2; ++sm)
#pragma unroll
    for (int sn = 0; sn < 2; ++sn)
#pragma unroll
      for (int r = 0; r < 8; ++r) {
        int R = rowBase + wm * 32 + sm * 16 + r + 8 * hi;
        int C = colBase + wn * 32 + sn * 16 + l15;
        float val = (acc[sm][sn][r] + bias[C]) * scale;
        int b = R >> 11, s = R & (S_ - 1), h = C >> 6, d = C & (HD_ - 1);
        size_t idx = vmode
            ? ((((size_t)(b * H_ + h)) * HD_ + d) * S_ + s)   // V: [B,H,hd,S]
            : ((((size_t)(b * H_ + h)) * S_ + s) * HD_ + d);  // Q/K: [B,H,S,hd]
        dst[idx] = (half_t)val;
      }
}

// ---------------------------------------------------------------------------
// Output projection: out(f32) = Oh(f16) @ Wf(f16, pre-transposed) + bf
// ---------------------------------------------------------------------------
__global__ void out_gemm_kernel(const half_t* __restrict__ Oh,
                                const half_t* __restrict__ Wft,
                                const float* __restrict__ bf,
                                float* __restrict__ out) {
  __shared__ __align__(16) half_t As[64][48];
  __shared__ __align__(16) half_t Bt[128][48];
  const int rowBase = blockIdx.y * 64;
  const int colBase = blockIdx.x * 128;
  v8f acc[2][2] = {};
  gemm_tile_core(Oh, Wft, As, Bt, rowBase, colBase, acc);

  const int lane = threadIdx.x & 31;
  const int wid  = threadIdx.x >> 5;
  const int wm = wid >> 2, wn = wid & 3, hi = lane >> 4, l15 = lane & 15;
#pragma unroll
  for (int sm = 0; sm < 2; ++sm)
#pragma unroll
    for (int sn = 0; sn < 2; ++sn)
#pragma unroll
      for (int r = 0; r < 8; ++r) {
        int R = rowBase + wm * 32 + sm * 16 + r + 8 * hi;
        int C = colBase + wn * 32 + sn * 16 + l15;
        out[(size_t)R * D_ + C] = acc[sm][sn][r] + bf[C];
      }
}

// ---------------------------------------------------------------------------
// Flash attention with bidirectional ALiBi.
// Block = 256 threads (8 waves) covers 128 q rows of one (b, h).
// Each wave owns a 16-row q tile; K (row-major) and V (already transposed in
// global) staged in LDS per 32-key chunk with b128-only traffic. Chunk range
// pruned by head direction; finite -1e9 mask matches the reference.
// ---------------------------------------------------------------------------
__global__ void attn_kernel(const half_t* __restrict__ Qh,
                            const half_t* __restrict__ Kh,
                            const half_t* __restrict__ VhT,
                            half_t* __restrict__ Oh) {
  __shared__ __align__(16) half_t Kt[32][72];      // row-major [key][dim]
  __shared__ __align__(16) half_t Vt[64][48];      // [dim][key] (pre-transposed)
  __shared__ __align__(16) half_t Pst[8][16][32];  // per-wave P re-layout

  const int bh  = blockIdx.x;
  const int b   = bh >> 4;
  const int h   = bh & 15;
  const int Qb  = blockIdx.y * 128;
  const int tid = threadIdx.x;
  const int lane = tid & 31;
  const int wid  = tid >> 5;
  const int hi   = lane >> 4;
  const int l15  = lane & 15;
  const int qbase = Qb + wid * 16;

  const half_t* Qp = Qh  + (size_t)(b * H_ + h) * S_ * HD_;
  const half_t* Kp = Kh  + (size_t)(b * H_ + h) * S_ * HD_;
  const half_t* Vp = VhT + (size_t)(b * H_ + h) * HD_ * S_;  // [dim][key]

  // Q fragments for the whole hd=64 (two K=32 steps), loaded once.
  V16 aq0, aq1;
  {
    const half_t* qrow = Qp + (size_t)(qbase + l15) * HD_;
    aq0.h[0] = *(const v8h*)(qrow + hi * 8);
    aq0.h[1] = *(const v8h*)(qrow + 16 + hi * 8);
    aq1.h[0] = *(const v8h*)(qrow + 32 + hi * 8);
    aq1.h[1] = *(const v8h*)(qrow + 48 + hi * 8);
  }

  float M[8], L[8];
  v8f acc[4] = {};
#pragma unroll
  for (int r = 0; r < 8; ++r) { M[r] = -1e30f; L[r] = 0.0f; }

  // slope = (24^(1/8))^-(h%8+1)
  const float slope = __expf(-__logf(24.0f) / 8.0f * (float)((h & 7) + 1));
  const bool fwd = (h < 8);
  const int cBeg = fwd ? 0 : (Qb >> 5);
  const int cEnd = fwd ? ((Qb >> 5) + 4) : (S_ >> 5);

  for (int kc = cBeg; kc < cEnd; ++kc) {
    const int kbase = kc * 32;
    // ---- stage K (row-major) and V (transposed in global): b128 only -----
    {
      int key = tid >> 3, db = (tid & 7) * 8;           // K: 32 keys x 64 dims
      v8h kv = *(const v8h*)&Kp[(size_t)(kbase + key) * HD_ + db];
      *(v8h*)&Kt[key][db] = kv;
      int d = tid >> 2, k8 = (tid & 3) * 8;             // V: 64 dims x 32 keys
      v8h vv = *(const v8h*)&Vp[(size_t)d * S_ + kbase + k8];
      *(v8h*)&Vt[d][k8] = vv;
    }
    __syncthreads();

    // ---- scores: Q(16x64) x K^T(64x32) -> two 16x16 f32 tiles ------------
    v8f c0 = {}, c1 = {};
    {
      V16 bk0, bk0b, bk1, bk1b;
      int key0 = l15, key1 = 16 + l15;
      bk0.h[0]  = *(const v8h*)&Kt[key0][hi * 8];
      bk0.h[1]  = *(const v8h*)&Kt[key0][16 + hi * 8];
      bk0b.h[0] = *(const v8h*)&Kt[key0][32 + hi * 8];
      bk0b.h[1] = *(const v8h*)&Kt[key0][48 + hi * 8];
      bk1.h[0]  = *(const v8h*)&Kt[key1][hi * 8];
      bk1.h[1]  = *(const v8h*)&Kt[key1][16 + hi * 8];
      bk1b.h[0] = *(const v8h*)&Kt[key1][32 + hi * 8];
      bk1b.h[1] = *(const v8h*)&Kt[key1][48 + hi * 8];
      c0 = __builtin_amdgcn_wmma_f32_16x16x32_f16(false, aq0.v, false, bk0.v,
                                                  (short)0, c0, false, false);
      c0 = __builtin_amdgcn_wmma_f32_16x16x32_f16(false, aq1.v, false, bk0b.v,
                                                  (short)0, c0, false, false);
      c1 = __builtin_amdgcn_wmma_f32_16x16x32_f16(false, aq0.v, false, bk1.v,
                                                  (short)0, c1, false, false);
      c1 = __builtin_amdgcn_wmma_f32_16x16x32_f16(false, aq1.v, false, bk1b.v,
                                                  (short)0, c1, false, false);
    }

    // ---- ALiBi bias + online softmax (per-row, 16-lane half reductions) --
    float p0[8], p1[8];
#pragma unroll
    for (int r = 0; r < 8; ++r) {
      int iq = qbase + r + 8 * hi;
      int j0 = kbase + l15;
      int j1 = kbase + 16 + l15;
      bool a0 = fwd ? (j0 <= iq) : (j0 >= iq);
      bool a1 = fwd ? (j1 <= iq) : (j1 >= iq);
      float s0 = c0[r] + (a0 ? -slope * fabsf((float)(iq - j0)) : -1e9f);
      float s1 = c1[r] + (a1 ? -slope * fabsf((float)(iq - j1)) : -1e9f);
      float mx = fmaxf(s0, s1);
#pragma unroll
      for (int o = 8; o >= 1; o >>= 1) mx = fmaxf(mx, __shfl_xor(mx, o, 32));
      float Mn = fmaxf(M[r], mx);
      float sc = __expf(M[r] - Mn);
      p0[r] = __expf(s0 - Mn);
      p1[r] = __expf(s1 - Mn);
      float rs = p0[r] + p1[r];
#pragma unroll
      for (int o = 8; o >= 1; o >>= 1) rs += __shfl_xor(rs, o, 32);
      L[r] = L[r] * sc + rs;
      M[r] = Mn;
#pragma unroll
      for (int j = 0; j < 4; ++j) acc[j][r] = acc[j][r] * sc;
    }

    // ---- re-layout P (C layout -> A layout) through per-wave LDS ---------
    half_t (*pp)[32] = Pst[wid];
#pragma unroll
    for (int r = 0; r < 8; ++r) {
      pp[r + 8 * hi][l15]      = (half_t)p0[r];
      pp[r + 8 * hi][16 + l15] = (half_t)p1[r];
    }
    asm volatile("s_wait_dscnt 0x0" ::: "memory");  // wave-local store->load
    V16 pf;
    pf.h[0] = *(const v8h*)&pp[l15][hi * 8];
    pf.h[1] = *(const v8h*)&pp[l15][16 + hi * 8];

    // ---- P(16x32) x V(32x64): 4 WMMAs accumulate into output -------------
#pragma unroll
    for (int j = 0; j < 4; ++j) {
      V16 bv;
      int dcol = j * 16 + l15;
      bv.h[0] = *(const v8h*)&Vt[dcol][hi * 8];
      bv.h[1] = *(const v8h*)&Vt[dcol][16 + hi * 8];
      acc[j] = __builtin_amdgcn_wmma_f32_16x16x32_f16(false, pf.v, false, bv.v,
                                                      (short)0, acc[j],
                                                      false, false);
    }
    __syncthreads();
  }

  // ---- epilogue: normalize and write f16 into [B*S, D] for final GEMM ----
#pragma unroll
  for (int r = 0; r < 8; ++r) {
    float inv = 1.0f / L[r];
    int R = b * S_ + qbase + r + 8 * hi;
#pragma unroll
    for (int j = 0; j < 4; ++j) {
      Oh[(size_t)R * D_ + h * HD_ + j * 16 + l15] = (half_t)(acc[j][r] * inv);
    }
  }
}

// ---------------------------------------------------------------------------
// Host-side launcher. Workspace layout (all 16B-aligned, ~48 MB total):
//   xh   : ROWS*D f16      (8 MB)
//   wqt/wkt/wvt/wft : D*D f16 each, TRANSPOSED [n][k] (4 x 2 MB)
//   Qh/Kh : B*H*S*hd f16  (2 x 8 MB);  VhT : B*H*hd*S f16 (8 MB)
//   Oh   : ROWS*D f16      (8 MB)
// ---------------------------------------------------------------------------
extern "C" void kernel_launch(void* const* d_in, const int* in_sizes, int n_in,
                              void* d_out, int out_size, void* d_ws,
                              size_t ws_size, hipStream_t stream) {
  const float* x  = (const float*)d_in[0];
  const float* Wq = (const float*)d_in[1];
  const float* bq = (const float*)d_in[2];
  const float* Wk = (const float*)d_in[3];
  const float* bk = (const float*)d_in[4];
  const float* Wv = (const float*)d_in[5];
  const float* bv = (const float*)d_in[6];
  const float* Wf = (const float*)d_in[7];
  const float* bf = (const float*)d_in[8];
  float* out = (float*)d_out;

  char* ws = (char*)d_ws;
  size_t off = 0;
  half_t* xh  = (half_t*)(ws + off); off += (size_t)ROWS * D_ * sizeof(half_t);
  half_t* wqt = (half_t*)(ws + off); off += (size_t)D_ * D_ * sizeof(half_t);
  half_t* wkt = (half_t*)(ws + off); off += (size_t)D_ * D_ * sizeof(half_t);
  half_t* wvt = (half_t*)(ws + off); off += (size_t)D_ * D_ * sizeof(half_t);
  half_t* wft = (half_t*)(ws + off); off += (size_t)D_ * D_ * sizeof(half_t);
  half_t* Qh  = (half_t*)(ws + off); off += (size_t)B_ * H_ * S_ * HD_ * sizeof(half_t);
  half_t* Kh  = (half_t*)(ws + off); off += (size_t)B_ * H_ * S_ * HD_ * sizeof(half_t);
  half_t* VhT = (half_t*)(ws + off); off += (size_t)B_ * H_ * HD_ * S_ * sizeof(half_t);
  half_t* Oh  = (half_t*)(ws + off); off += (size_t)ROWS * D_ * sizeof(half_t);

  // 1) conversions: x plain; weights transpose-converted to [n][k]
  cvt_f32_to_f16<<<1024, 256, 0, stream>>>(x, xh, ROWS * D_ / 4);
  dim3 tgrid(D_ / 64, D_ / 64);
  cvt_transpose_f32_to_f16<<<tgrid, 256, 0, stream>>>(Wq, wqt);
  cvt_transpose_f32_to_f16<<<tgrid, 256, 0, stream>>>(Wk, wkt);
  cvt_transpose_f32_to_f16<<<tgrid, 256, 0, stream>>>(Wv, wvt);
  cvt_transpose_f32_to_f16<<<tgrid, 256, 0, stream>>>(Wf, wft);

  // 2) fused QKV projection (z selects Q/K/V), tiles 64x128, 8 waves/block
  qkv_gemm_kernel<<<dim3(D_ / 128, ROWS / 64, 3), 256, 0, stream>>>(
      xh, wqt, wkt, wvt, bq, bk, bv, Qh, Kh, VhT);

  // 3) flash attention with bidirectional ALiBi
  attn_kernel<<<dim3(B_ * H_, S_ / 128), 256, 0, stream>>>(Qh, Kh, VhT, Oh);

  // 4) output projection to f32
  out_gemm_kernel<<<dim3(D_ / 128, ROWS / 64), 256, 0, stream>>>(
      Oh, wft, bf, out);
}
